// SGFormerNet_25658134626479
// MI455X (gfx1250) — matrix operand order repaired and checked by reading, not compiled, not run on hardware
//
#include <hip/hip_runtime.h>
#include <hip/hip_bf16.h>

#define NNODES 50000
#define NEDGES 800000
#define CDIM   256
#define ODIM   128
#define GN_ROWS 128
#define EPS_ATT  1e-6f
#define EPS_NORM 1e-5f

#define NTROW   (NNODES / 16)              // 3125 row tiles
#define RGROUPS ((NTROW + 31) / 32)        // 98 blocks of 32 row-tiles

typedef __attribute__((ext_vector_type(16))) __bf16 v16bf;
typedef __attribute__((ext_vector_type(8)))  float  v8f;

// ---------------------------------------------------------------------------
// WMMA GEMM, bf16 inputs, f32 accumulate, LDS-staged weights.
//   A  : [M][K]    bf16 row-major
//   Wt : [Ncol][K] bf16 (pre-transposed)
// Block: 256 threads (8 waves). blockIdx.y selects a 16-column slice of Wt,
// staged once into LDS via global_load_async_to_lds_b128 (ASYNCcnt path) and
// reused by all 8 waves x 4 row-tiles (32 WMMA tiles per block).
// Each wave: 4 row-tiles x 1 col-tile, K fully unrolled.
// ---------------------------------------------------------------------------
template <int K, int NC, bool HB, bool HA, bool HR, bool RELU>
__global__ void gemm_bf16_wmma(const __bf16* __restrict__ A,
                               const __bf16* __restrict__ Wt,
                               const float* __restrict__ bias,
                               const float* addend,
                               const float* __restrict__ rowmul,
                               float* C, int M) {
  __shared__ __bf16 Bs[16 * K];            // 8 KB for K=256
  int tid  = threadIdx.x;
  int lane = tid & 31, wave = tid >> 5;
  int col0 = blockIdx.y * 16;
  int ntrow = M >> 4;

  // ---- async-stage Wt[col0..col0+15][0..K-1] into LDS (16*K*2 = 8192 B) ----
  {
    const char* src = (const char*)(Wt + (size_t)col0 * K);
    unsigned lds0 = (unsigned)(tid * 16);
    unsigned lds1 = lds0 + 4096u;
    const char* g0 = src + tid * 16;
    const char* g1 = src + tid * 16 + 4096;
    asm volatile("global_load_async_to_lds_b128 %0, %1, off"
                 :: "v"(lds0), "v"(g0) : "memory");
    asm volatile("global_load_async_to_lds_b128 %0, %1, off"
                 :: "v"(lds1), "v"(g1) : "memory");
    asm volatile("s_wait_asynccnt 0x0" ::: "memory");
  }
  __syncthreads();

  int m = lane & 15, half = lane >> 4;
  int base = blockIdx.x * 32 + wave * 4;   // first of this wave's 4 row-tiles

  bool valid[4];
  const __bf16* arow[4];
#pragma unroll
  for (int j = 0; j < 4; ++j) {
    valid[j] = (base + j) < ntrow;         // wave-uniform
    arow[j]  = A + (size_t)((base + j) * 16 + m) * K;
    if (valid[j]) __builtin_prefetch(arow[j], 0, 0);   // global_prefetch_b8
  }

  v8f acc[4];
#pragma unroll
  for (int j = 0; j < 4; ++j) acc[j] = (v8f){0.f,0.f,0.f,0.f,0.f,0.f,0.f,0.f};

#pragma unroll
  for (int k0 = 0; k0 < K; k0 += 32) {
    // B 32x16 fragment from LDS: lanes 0-15 K=0..15, lanes 16-31 K=16..31
    v16bf b;
    const __bf16* bp = &Bs[m * K + k0 + half * 16];
#pragma unroll
    for (int i = 0; i < 16; ++i) b[i] = bp[i];
#pragma unroll
    for (int j = 0; j < 4; ++j) {
      if (!valid[j]) continue;             // wave-uniform branch, EXEC intact
      v16bf a;
      const __bf16* ap = arow[j] + k0 + half * 16;
#pragma unroll
      for (int i = 0; i < 8; ++i) { a[i] = ap[i]; a[8 + i] = ap[8 + i]; }
      acc[j] = __builtin_amdgcn_wmma_f32_16x16x32_bf16(false, a, false, b,
                                                       (short)0, acc[j], false, false);
    }
  }

  int mo = half * 8;
#pragma unroll
  for (int j = 0; j < 4; ++j) {
    if (!valid[j]) continue;
    int row0 = (base + j) * 16;
#pragma unroll
    for (int r = 0; r < 8; ++r) {
      int row = row0 + mo + r, col = col0 + m;
      float v = acc[j][r];
      if (HR)   v *= rowmul[row];
      if (HA)   v += addend[(size_t)row * NC + col];
      if (HB)   v += bias[col];
      if (RELU) v  = fmaxf(v, 0.f);
      C[(size_t)row * NC + col] = v;
    }
  }
}

// ---------------------------------------------------------------------------
// KV = K^T @ V (CDIM x CDIM), WMMA reduction over N, f32 atomic accumulation.
// ---------------------------------------------------------------------------
__global__ void ktv_wmma(const float* __restrict__ Kmat,
                         const float* __restrict__ Vmat,
                         float* KV, int n) {
  int lane = threadIdx.x & 31, wave = threadIdx.x >> 5;
  int ti = blockIdx.x >> 4, tj = blockIdx.x & 15;     // d=256 -> 16 col tiles
  int m = lane & 15, half = lane >> 4;

  v8f acc = {0.f,0.f,0.f,0.f,0.f,0.f,0.f,0.f};
  int colA = ti * 16 + m;
  int colB = tj * 16 + m;

  for (int n0 = wave * 32; n0 < n; n0 += 32 * 8) {
    v16bf a, b;
#pragma unroll
    for (int i = 0; i < 8; ++i) {
      int r = n0 + half * 8 + i;
      a[i] = (__bf16)(r < n ? Kmat[(size_t)r * CDIM + colA] : 0.f);
    }
#pragma unroll
    for (int i = 0; i < 8; ++i) {
      int r = n0 + 16 + half * 8 + i;
      a[8 + i] = (__bf16)(r < n ? Kmat[(size_t)r * CDIM + colA] : 0.f);
    }
#pragma unroll
    for (int i = 0; i < 16; ++i) {
      int r = n0 + (i < 8 ? half * 8 + i : 16 + half * 8 + (i - 8));
      b[i] = (__bf16)(r < n ? Vmat[(size_t)r * CDIM + colB] : 0.f);
    }
    acc = __builtin_amdgcn_wmma_f32_16x16x32_bf16(false, a, false, b,
                                                  (short)0, acc, false, false);
  }
  int mo = half * 8;
#pragma unroll
  for (int r = 0; r < 8; ++r)
    atomicAdd(&KV[(size_t)(ti * 16 + mo + r) * CDIM + tj * 16 + m], acc[r]);
}

// ---------------------------------------------------------------------------
__global__ void f32_to_bf16(const float* __restrict__ src, __bf16* dst, size_t n) {
  size_t t = (size_t)blockIdx.x * blockDim.x + threadIdx.x;
  if (t < n) dst[t] = (__bf16)src[t];
}

// Wt[n][k] = (bf16) W[k][n]
__global__ void w_to_bf16t(const float* __restrict__ W, __bf16* Wt, int K, int Ncol) {
  int tid = blockIdx.x * blockDim.x + threadIdx.x;
  if (tid >= K * Ncol) return;
  int k = tid / Ncol, n = tid % Ncol;
  Wt[(size_t)n * K + k] = (__bf16)W[tid];
}

__global__ void fillf(float* p, float v, size_t n) {
  size_t t = (size_t)blockIdx.x * blockDim.x + threadIdx.x;
  if (t < n) p[t] = v;
}

__global__ void deg_scatter(const int* __restrict__ dst, float* deg, int e) {
  int t = blockIdx.x * blockDim.x + threadIdx.x;
  if (t < e) atomicAdd(&deg[dst[t]], 1.0f);
}

__global__ void rsqrt_inplace(float* p, int n) {
  int t = blockIdx.x * blockDim.x + threadIdx.x;
  if (t < n) p[t] = rsqrtf(p[t]);
}

__global__ void agg_init(const float* __restrict__ h, const float* __restrict__ dinv,
                         const float* __restrict__ bias, float* out, int n) {
  size_t tid = (size_t)blockIdx.x * blockDim.x + threadIdx.x;
  if (tid >= (size_t)n * CDIM) return;
  int i = (int)(tid >> 8), c = (int)(tid & 255);
  float di = dinv[i];
  out[tid] = h[tid] * di * di + bias[c];
}

// one wave per edge: uniform index loads (scalar), coalesced channel traffic
__global__ void gcn_scatter(const float* __restrict__ h, const float* __restrict__ dinv,
                            const int* __restrict__ src, const int* __restrict__ dst,
                            float* out, int e) {
  int lane = threadIdx.x & 31, wave = threadIdx.x >> 5;
  int eidx = blockIdx.x * 8 + wave;
  if (eidx >= e) return;                   // wave-uniform
  int s = src[eidx], t = dst[eidx];
  float coef = dinv[s] * dinv[t];
  const float4* hp = (const float4*)(h + (size_t)s * CDIM);
  float4 h0 = hp[lane * 2], h1 = hp[lane * 2 + 1];
  float* op = out + (size_t)t * CDIM + lane * 8;
  atomicAdd(op + 0, h0.x * coef);
  atomicAdd(op + 1, h0.y * coef);
  atomicAdd(op + 2, h0.z * coef);
  atomicAdd(op + 3, h0.w * coef);
  atomicAdd(op + 4, h1.x * coef);
  atomicAdd(op + 5, h1.y * coef);
  atomicAdd(op + 6, h1.z * coef);
  atomicAdd(op + 7, h1.w * coef);
}

__global__ void colsum(const float* __restrict__ x, float* sums, int n) {
  int c = threadIdx.x;
  int i0 = blockIdx.x * GN_ROWS;
  int i1 = i0 + GN_ROWS; if (i1 > n) i1 = n;
  float s = 0.f;
  for (int i = i0; i < i1; ++i) s += x[(size_t)i * CDIM + c];
  atomicAdd(&sums[c], s);
}

__global__ void gn_center(const float* __restrict__ x, const float* __restrict__ sums,
                          const float* __restrict__ mscale, float* centered,
                          float* varsum, int n) {
  int c = threadIdx.x;
  int i0 = blockIdx.x * GN_ROWS;
  int i1 = i0 + GN_ROWS; if (i1 > n) i1 = n;
  float mean = sums[c] * (1.0f / n);
  float ms   = mscale[c];
  float sq = 0.f;
  for (int i = i0; i < i1; ++i) {
    float v = x[(size_t)i * CDIM + c] - ms * mean;
    centered[(size_t)i * CDIM + c] = v;
    sq += v * v;
  }
  atomicAdd(&varsum[c], sq);
}

__global__ void gn_final(const float* __restrict__ centered, const float* __restrict__ varsum,
                         const float* __restrict__ w, const float* __restrict__ b,
                         const float* residual, float* out, int n) {
  size_t tid = (size_t)blockIdx.x * blockDim.x + threadIdx.x;
  if (tid >= (size_t)n * CDIM) return;
  int c = (int)(tid & 255);
  float rs = rsqrtf(varsum[c] * (1.0f / n) + EPS_NORM);
  float v = w[c] * centered[tid] * rs + b[c];
  v = fmaxf(v, 0.f);
  if (residual) v += residual[tid];
  out[tid] = v;
}

// stores RECIPROCAL normalizer: 1 / (dot(Q[i],ksum) + eps)
__global__ void rowdot_rcp(const float* __restrict__ Q, const float* __restrict__ ksum,
                           float* normv, int n) {
  int lane = threadIdx.x & 31, wave = threadIdx.x >> 5;
  int row = blockIdx.x * 8 + wave;
  if (row >= n) return;
  float s = 0.f;
  for (int c = lane; c < CDIM; c += 32) s += Q[(size_t)row * CDIM + c] * ksum[c];
#pragma unroll
  for (int off = 16; off; off >>= 1) s += __shfl_xor(s, off);
  if (lane == 0) normv[row] = 1.0f / (s + EPS_ATT);
}

// ---------------------------------------------------------------------------
extern "C" void kernel_launch(void* const* d_in, const int* in_sizes, int n_in,
                              void* d_out, int out_size, void* d_ws, size_t ws_size,
                              hipStream_t stream) {
  (void)in_sizes; (void)n_in; (void)out_size; (void)ws_size;
  const float* x   = (const float*)d_in[0];
  const int*   ei  = (const int*)d_in[1];
  const int* esrc  = ei;
  const int* edst  = ei + NEDGES;
  const float* c1W = (const float*)d_in[2];
  const float* c1b = (const float*)d_in[3];
  const float* qW  = (const float*)d_in[4];
  const float* qb  = (const float*)d_in[5];
  const float* kW  = (const float*)d_in[6];
  const float* kb  = (const float*)d_in[7];
  const float* vW  = (const float*)d_in[8];
  const float* vb  = (const float*)d_in[9];
  const float* n1w = (const float*)d_in[10];
  const float* n1b = (const float*)d_in[11];
  const float* n1m = (const float*)d_in[12];
  const float* c2W = (const float*)d_in[13];
  const float* c2b = (const float*)d_in[14];
  const float* n2w = (const float*)d_in[21];
  const float* n2b = (const float*)d_in[22];
  const float* n2m = (const float*)d_in[23];
  const float* c3W = (const float*)d_in[24];
  const float* c3b = (const float*)d_in[25];
  const float* n3w = (const float*)d_in[32];
  const float* n3b = (const float*)d_in[33];
  const float* n3m = (const float*)d_in[34];
  const float* hW  = (const float*)d_in[35];
  const float* hb  = (const float*)d_in[36];
  float* out = (float*)d_out;

  float* P = (float*)d_ws;
  const size_t NB = (size_t)NNODES * CDIM;
  float* B0 = P;            float* B1 = P + NB;     float* B2 = P + 2 * NB;
  float* B3 = P + 3 * NB;   float* B4 = P + 4 * NB;
  float* dinv   = P + 5 * NB;
  float* normv  = dinv + NNODES;
  float* KV     = normv + NNODES;
  float* ksum   = KV + CDIM * CDIM;
  float* sums   = ksum + CDIM;
  float* varsum = sums + CDIM;
  __bf16* Ax    = (__bf16*)(varsum + CDIM);
  __bf16* Abuf  = Ax + NB;
  __bf16* Wt    = Abuf + NB;

  const int TB = 256;
  auto cdiv = [](size_t a, size_t b) { return (int)((a + b - 1) / b); };
  const dim3 G256(RGROUPS, CDIM / 16);    // 98 x 16
  const dim3 G128(RGROUPS, ODIM / 16);    // 98 x 8

  auto convA = [&](const float* src, __bf16* dst) {
    f32_to_bf16<<<cdiv(NB, TB), TB, 0, stream>>>(src, dst, NB);
  };
  auto convW = [&](const float* W, int K, int Ncol) {
    w_to_bf16t<<<cdiv((size_t)K * Ncol, TB), TB, 0, stream>>>(W, Wt, K, Ncol);
  };
  auto graphnorm = [&](const float* xin, const float* w, const float* b,
                       const float* ms, const float* res, float* scratch, float* y) {
    fillf<<<cdiv(2 * CDIM, TB), TB, 0, stream>>>(sums, 0.f, 2 * CDIM);
    colsum<<<cdiv(NNODES, GN_ROWS), CDIM, 0, stream>>>(xin, sums, NNODES);
    gn_center<<<cdiv(NNODES, GN_ROWS), CDIM, 0, stream>>>(xin, sums, ms, scratch,
                                                          varsum, NNODES);
    gn_final<<<cdiv(NB, TB), TB, 0, stream>>>(scratch, varsum, w, b, res, y, NNODES);
  };
  auto gcn_agg = [&](const float* h, const float* bias, float* dstbuf) {
    agg_init<<<cdiv(NB, TB), TB, 0, stream>>>(h, dinv, bias, dstbuf, NNODES);
    gcn_scatter<<<cdiv(NEDGES, 8), TB, 0, stream>>>(h, dinv, esrc, edst, dstbuf, NEDGES);
  };

  // degree -> dinv
  fillf<<<cdiv(NNODES, TB), TB, 0, stream>>>(dinv, 1.0f, NNODES);
  deg_scatter<<<cdiv(NEDGES, TB), TB, 0, stream>>>(edst, dinv, NEDGES);
  rsqrt_inplace<<<cdiv(NNODES, TB), TB, 0, stream>>>(dinv, NNODES);

  convA(x, Ax);

  // ---- layer 1 ----
  convW(c1W, CDIM, CDIM);
  gemm_bf16_wmma<CDIM, CDIM, false, false, false, false>
      <<<G256, TB, 0, stream>>>(Ax, Wt, nullptr, nullptr, nullptr, B0, NNODES);  // h1
  gcn_agg(B0, c1b, B1);                                                          // x1_g
  convW(qW, CDIM, CDIM);
  gemm_bf16_wmma<CDIM, CDIM, true, false, false, true>
      <<<G256, TB, 0, stream>>>(Ax, Wt, qb, nullptr, nullptr, B2, NNODES);       // Q
  convW(kW, CDIM, CDIM);
  gemm_bf16_wmma<CDIM, CDIM, true, false, false, true>
      <<<G256, TB, 0, stream>>>(Ax, Wt, kb, nullptr, nullptr, B3, NNODES);       // K
  convW(vW, CDIM, CDIM);
  gemm_bf16_wmma<CDIM, CDIM, true, false, false, false>
      <<<G256, TB, 0, stream>>>(Ax, Wt, vb, nullptr, nullptr, B4, NNODES);       // V
  fillf<<<cdiv(CDIM * CDIM + CDIM, TB), TB, 0, stream>>>(KV, 0.f, CDIM * CDIM + CDIM);
  ktv_wmma<<<(CDIM / 16) * (CDIM / 16), TB, 0, stream>>>(B3, B4, KV, NNODES);
  colsum<<<cdiv(NNODES, GN_ROWS), CDIM, 0, stream>>>(B3, ksum, NNODES);
  rowdot_rcp<<<cdiv(NNODES, 8), TB, 0, stream>>>(B2, ksum, normv, NNODES);
  convA(B2, Abuf);
  convW(KV, CDIM, CDIM);
  gemm_bf16_wmma<CDIM, CDIM, false, true, true, false>
      <<<G256, TB, 0, stream>>>(Abuf, Wt, nullptr, B1, normv, B0, NNODES);       // x1_g+attn
  graphnorm(B0, n1w, n1b, n1m, nullptr, B2, B1);                                 // B1 = x1

  // ---- layer 2 ----
  convA(B1, Abuf);
  convW(c2W, CDIM, CDIM);
  gemm_bf16_wmma<CDIM, CDIM, false, false, false, false>
      <<<G256, TB, 0, stream>>>(Abuf, Wt, nullptr, nullptr, nullptr, B0, NNODES);
  gcn_agg(B0, c2b, B2);
  graphnorm(B2, n2w, n2b, n2m, B1, B3, B4);                                      // B4 = x2

  // ---- layer 3 ----
  convA(B4, Abuf);
  convW(c3W, CDIM, CDIM);
  gemm_bf16_wmma<CDIM, CDIM, false, false, false, false>
      <<<G256, TB, 0, stream>>>(Abuf, Wt, nullptr, nullptr, nullptr, B0, NNODES);
  gcn_agg(B0, c3b, B2);
  graphnorm(B2, n3w, n3b, n3m, B4, B3, B0);                                      // B0 = x3

  // ---- head: out = [x, x3] @ hW + hb ----
  convW(hW, CDIM, ODIM);
  gemm_bf16_wmma<CDIM, ODIM, true, false, false, false>
      <<<G128, TB, 0, stream>>>(Ax, Wt, hb, nullptr, nullptr, out, NNODES);
  convA(B0, Abuf);
  convW(hW + (size_t)CDIM * ODIM, CDIM, ODIM);
  gemm_bf16_wmma<CDIM, ODIM, false, true, false, false>
      <<<G128, TB, 0, stream>>>(Abuf, Wt, nullptr, out, nullptr, out, NNODES);
}
